// RelationalMemoryCoreCell_17265768530036
// MI455X (gfx1250) — matrix operand order repaired and checked by reading, not compiled, not run on hardware
//
#include <hip/hip_runtime.h>
#include <hip/hip_bf16.h>
#include <math.h>

// Problem constants (from reference)
#define Bm 4096
#define Sm 8
#define Dm 1024
#define Hm 16
#define Fm 4096
#define Gm 2048
#define Tm 9               // S+1
#define NROWS (Bm * Tm)    // 36864
#define QKV_N (3 * Dm)     // fused q|k|v output width

typedef __attribute__((ext_vector_type(16))) __bf16        v16bf;
typedef __attribute__((ext_vector_type(8)))  float         v8f;
typedef __attribute__((ext_vector_type(4)))  unsigned int  v4u;

struct Frag2 { v4u lo, hi; };   // 32B -> bit_cast to v16bf

__device__ __forceinline__ unsigned short f2bf(float f) {
  unsigned u = __builtin_bit_cast(unsigned, f);
  u += 0x7fffu + ((u >> 16) & 1u);   // round-to-nearest-even
  return (unsigned short)(u >> 16);
}

// ---------------- elementwise convert f32 -> bf16 (optional tanh) ----------------
__global__ void cvt_bf16_kernel(const float* __restrict__ in,
                                unsigned short* __restrict__ out,
                                long n, int do_tanh) {
  long i = (long)blockIdx.x * blockDim.x + threadIdx.x;
  if (i >= n) return;
  float v = in[i];
  if (do_tanh) v = tanhf(v);
  out[i] = f2bf(v);
}

__global__ void copy_f32_kernel(const float* __restrict__ src,
                                float* __restrict__ dst, int n) {
  int i = blockIdx.x * blockDim.x + threadIdx.x;
  if (i < n) dst[i] = src[i];
}

// ---- weight convert+pack: W (K x Nw f32, Nw = 1<<nshift) into dword matrix -----
// out[k2*outN + outOff + n] = bf16(W[2k2,n]) | bf16(W[2k2+1,n])<<16 (WMMA K-pairs)
__global__ void pack_weight_kernel(const float* __restrict__ W,
                                   unsigned* __restrict__ out,
                                   long n_items, int nshift, int outN, int outOff) {
  long i = (long)blockIdx.x * blockDim.x + threadIdx.x;
  if (i >= n_items) return;
  long k2 = i >> nshift;
  long n  = i & ((1L << nshift) - 1);
  long Nw = 1L << nshift;
  unsigned lo = f2bf(W[(2 * k2)     * Nw + n]);
  unsigned hi = f2bf(W[(2 * k2 + 1) * Nw + n]);
  out[k2 * outN + outOff + n] = lo | (hi << 16);
}

// ---------------- scatter memory rows into seq (rows 0..7 of each group of 9) ----
__global__ void mem_to_seq_kernel(const float* __restrict__ mem,
                                  float* __restrict__ seq) {
  long i = (long)blockIdx.x * blockDim.x + threadIdx.x;
  if (i >= (long)Bm * Sm * Dm) return;
  int d   = (int)(i & (Dm - 1));
  int row = (int)(i >> 10);        // b*8+s
  int b   = row >> 3;
  int s   = row & 7;
  seq[((long)(b * Tm + s) << 10) + d] = mem[i];
}

// ---------------- LayerNorm over D=1024, bf16 output ----------------------------
__global__ __launch_bounds__(256) void layernorm_bf16_kernel(
    const float* __restrict__ x, const float* __restrict__ g,
    const float* __restrict__ bta, unsigned short* __restrict__ out) {
  __shared__ float red0[256];
  __shared__ float red1[256];
  int row = blockIdx.x;
  const float* xr = x + (long)row * Dm;
  float v[4];
  float s = 0.f, sq = 0.f;
#pragma unroll
  for (int i = 0; i < 4; ++i) {
    v[i] = xr[threadIdx.x + i * 256];
    s += v[i]; sq += v[i] * v[i];
  }
  red0[threadIdx.x] = s; red1[threadIdx.x] = sq;
  __syncthreads();
  for (int off = 128; off > 0; off >>= 1) {
    if ((int)threadIdx.x < off) {
      red0[threadIdx.x] += red0[threadIdx.x + off];
      red1[threadIdx.x] += red1[threadIdx.x + off];
    }
    __syncthreads();
  }
  float mu  = red0[0] * (1.f / Dm);
  float var = red1[0] * (1.f / Dm) - mu * mu;
  float rst = rsqrtf(var + 1e-5f);
#pragma unroll
  for (int i = 0; i < 4; ++i) {
    int c = threadIdx.x + i * 256;
    out[(long)row * Dm + c] = f2bf((v[i] - mu) * rst * g[c] + bta[c]);
  }
}

// ---------------- Tiled bf16 WMMA GEMM: C = act(A@B + bias) (+Res) --------------
// N, K are template constants so all per-`it` tile-load offsets fold into the
// 24-bit instruction offset field (A: 16*it*Kw dwords <= 1.83MB; B: 2*it*N
// dwords <= 229KB) — no per-iteration 64-bit address chains.
// Block tile 128x128, K-step 32; 8 waves in 4(M)x2(N), each wave 32x64 = 8 WMMAs.
// LDS pitch 20 dwords (80B): every fragment half is one aligned ds_load_b128.
// Global->LDS is register-staged and software-pipelined.
template <int ACT, bool HAS_RES, bool OUT_BF, int N, int K>
__global__ __launch_bounds__(256) void gemm_wmma_bf16_kernel(
    const unsigned short* __restrict__ A, const unsigned* __restrict__ Bp,
    const float* __restrict__ bias, const float* __restrict__ Res,
    float* __restrict__ Cf, unsigned short* __restrict__ Cb,
    long c_base, int c_pitch) {
  __shared__ unsigned Asl[128 * 20];   // [row][k-pair dword], pitch 20
  __shared__ unsigned Bsl[128 * 20];   // transposed: [n][k2], pitch 20
  constexpr int Kw = K >> 1;           // A row length in dwords
  const unsigned* Au = (const unsigned*)A;
  const int tid  = threadIdx.x;
  const int lane = tid & 31;
  const int wid  = tid >> 5;
  const int wm   = wid & 3;            // M sub-block (32 rows)
  const int wn   = wid >> 2;           // N sub-block (64 cols)
  const int m0   = blockIdx.y * 128;
  const int n0   = blockIdx.x * 128;

  // per-thread global / LDS bases (per-`it` offsets are compile-time immediates)
  const int arow = tid >> 4, acol = tid & 15;   // A: 16 rows x 16 dwords per pass
  const int bn   = tid & 127, bk2 = tid >> 7;   // B: 2 k2-rows x 128 cols per pass
  const unsigned* aG = Au + (long)(m0 + arow) * Kw + acol;
  const unsigned* bG = Bp + (long)bk2 * N + n0 + bn;
  const int aLdsBase = arow * 20 + acol;
  const int bLdsBase = bn * 20 + bk2;

  v8f acc[2][4];
#pragma unroll
  for (int i = 0; i < 2; ++i)
#pragma unroll
    for (int j = 0; j < 4; ++j)
#pragma unroll
      for (int r = 0; r < 8; ++r) acc[i][j][r] = 0.f;

  unsigned areg[8], breg[8];
  auto load_tiles = [&](int k0) {
#pragma unroll
    for (int it = 0; it < 8; ++it)
      areg[it] = aG[(k0 >> 1) + 16 * it * Kw];          // imm offset: 16*it*Kw
#pragma unroll
    for (int it = 0; it < 8; ++it)
      breg[it] = bG[(long)(k0 >> 1) * N + 2 * it * N];  // imm offset: 2*it*N
  };

  const int lo16 = (lane < 16);
  const int l15  = lane & 15;

  load_tiles(0);
  for (int k0 = 0; k0 < K; k0 += 32) {
    __syncthreads();                    // prev iteration's LDS reads done
#pragma unroll
    for (int it = 0; it < 8; ++it) Asl[aLdsBase + it * 16 * 20] = areg[it];
#pragma unroll
    for (int it = 0; it < 8; ++it) Bsl[bLdsBase + it * 2] = breg[it];
    __syncthreads();
    if (k0 + 32 < K) load_tiles(k0 + 32);       // overlap with WMMAs below
    if (k0 + 64 < K) {                          // global_prefetch for tile after
      __builtin_prefetch(aG + ((k0 + 64) >> 1), 0, 3);
      __builtin_prefetch(bG + (long)((k0 + 64) >> 1) * N, 0, 3);
    }

    v16bf afr[2], bfr[4];
#pragma unroll
    for (int i = 0; i < 2; ++i) {
      // A frag: lanes<16 need dword quads {0-3,8-11}; lanes>=16 {4-7,12-15}
      const unsigned* base = &Asl[(wm * 32 + i * 16 + l15) * 20];
      Frag2 f;
      f.lo = *(const v4u*)(base + (lo16 ? 0 : 4));
      f.hi = *(const v4u*)(base + (lo16 ? 8 : 12));
      afr[i] = __builtin_bit_cast(v16bf, f);
    }
#pragma unroll
    for (int j2 = 0; j2 < 4; ++j2) {
      // B frag (transposed LDS): lanes<16 quads {0-3,8-11}; lanes>=16 {4-7,12-15}
      const unsigned* base = &Bsl[(wn * 64 + j2 * 16 + l15) * 20];
      Frag2 f;
      f.lo = *(const v4u*)(base + (lo16 ? 0 : 4));
      f.hi = *(const v4u*)(base + (lo16 ? 8 : 12));
      bfr[j2] = __builtin_bit_cast(v16bf, f);
    }
#pragma unroll
    for (int i = 0; i < 2; ++i)
#pragma unroll
      for (int j2 = 0; j2 < 4; ++j2)
        acc[i][j2] = __builtin_amdgcn_wmma_f32_16x16x32_bf16(
            false, afr[i], false, bfr[j2], (short)0, acc[i][j2], false, false);
  }

  // Epilogue (branch-free): VGPR r -> M=r (lanes<16) / M=8+r; lane&15 = N
  float bv[4];
#pragma unroll
  for (int j2 = 0; j2 < 4; ++j2)
    bv[j2] = bias[n0 + wn * 64 + j2 * 16 + l15];
#pragma unroll
  for (int i = 0; i < 2; ++i) {
#pragma unroll
    for (int j2 = 0; j2 < 4; ++j2) {
      int gcol = n0 + wn * 64 + j2 * 16 + l15;
#pragma unroll
      for (int r = 0; r < 8; ++r) {
        int grow = m0 + wm * 32 + i * 16 + (lo16 ? r : r + 8);
        long addr = c_base + (long)grow * c_pitch + gcol;
        float v = acc[i][j2][r] + bv[j2];
        if (ACT == 1) v = v > 0.f ? v : 0.f;
        if (HAS_RES) v += Res[addr];
        if (OUT_BF) Cb[addr] = f2bf(v);
        else        Cf[addr] = v;
      }
    }
  }
}

// -------- Attention on fused QKV buffer (row pitch 3072): one block per (b,h) ---
__global__ __launch_bounds__(64) void attention_kernel(
    const float* __restrict__ QKV, unsigned short* __restrict__ ctx_bf) {
  __shared__ float qs[Tm * 64], ks[Tm * 64], vs[Tm * 64], sc[Tm * Tm];
  int bh = blockIdx.x;
  int b  = bh >> 4;
  int h  = bh & 15;
  int tid = threadIdx.x;
  for (int idx = tid; idx < Tm * 64; idx += 64) {
    int t = idx >> 6, d = idx & 63;
    long base = (long)(b * Tm + t) * QKV_N + h * 64 + d;
    qs[idx] = QKV[base];
    ks[idx] = QKV[base + Dm];
    vs[idx] = QKV[base + 2 * Dm];
  }
  __syncthreads();
  for (int idx = tid; idx < Tm * Tm; idx += 64) {
    int i = idx / Tm, j = idx % Tm;
    float s = 0.f;
#pragma unroll
    for (int d = 0; d < 64; ++d) s += qs[i * 64 + d] * ks[j * 64 + d];
    sc[idx] = s * 0.125f;                       // 1/sqrt(64)
  }
  __syncthreads();
  if (tid < Tm) {
    float mx = -1e30f;
    for (int j = 0; j < Tm; ++j) mx = fmaxf(mx, sc[tid * Tm + j]);
    float sum = 0.f;
    for (int j = 0; j < Tm; ++j) { float e = __expf(sc[tid * Tm + j] - mx); sc[tid * Tm + j] = e; sum += e; }
    float inv = 1.f / sum;
    for (int j = 0; j < Tm; ++j) sc[tid * Tm + j] *= inv;
  }
  __syncthreads();
  for (int idx = tid; idx < Tm * 64; idx += 64) {
    int t = idx >> 6, d = idx & 63;
    float s = 0.f;
#pragma unroll
    for (int j = 0; j < Tm; ++j) s += sc[t * Tm + j] * vs[j * 64 + d];
    ctx_bf[((long)(b * Tm + t) << 10) + h * 64 + d] = f2bf(s);
  }
}

// ---------------- Final gate combine -> next_memory (written twice) -------------
__global__ void gate_combine_kernel(const float* __restrict__ enc,
                                    const float* __restrict__ mem,
                                    const float* __restrict__ gmem,
                                    const float* __restrict__ gin,
                                    float* __restrict__ out) {
  const long total = (long)Bm * Sm * Dm;
  long i = (long)blockIdx.x * blockDim.x + threadIdx.x;
  if (i >= total) return;
  int d   = (int)(i & (Dm - 1));
  int row = (int)(i >> 10);                    // b*8+s
  int b   = row >> 3;
  int s   = row & 7;
  float igp = gmem[(long)row * Gm + d]        + gin[(long)b * Gm + d];
  float fgp = gmem[(long)row * Gm + Dm + d]   + gin[(long)b * Gm + Dm + d] + 1.0f;
  float ig = 1.f / (1.f + __expf(-igp));
  float fg = 1.f / (1.f + __expf(-fgp));
  float e  = enc[((long)(b * Tm + s) << 10) + d];
  float v  = ig * tanhf(e) + fg * tanhf(mem[i]);
  out[i] = v;
  out[total + i] = v;
}

extern "C" void kernel_launch(void* const* d_in, const int* in_sizes, int n_in,
                              void* d_out, int out_size, void* d_ws, size_t ws_size,
                              hipStream_t stream) {
  (void)in_sizes; (void)n_in; (void)out_size; (void)ws_size;
  const float* x       = (const float*)d_in[0];
  const float* memory  = (const float*)d_in[1];
  const float* W_embed = (const float*)d_in[2];
  const float* b_embed = (const float*)d_in[3];
  const float* ln1_g   = (const float*)d_in[4];
  const float* ln1_b   = (const float*)d_in[5];
  const float* Wq = (const float*)d_in[6];  const float* bq = (const float*)d_in[7];
  const float* Wk = (const float*)d_in[8];  const float* bk = (const float*)d_in[9];
  const float* Wv = (const float*)d_in[10]; const float* bv = (const float*)d_in[11];
  const float* Wo = (const float*)d_in[12]; const float* bo = (const float*)d_in[13];
  const float* ln2_g = (const float*)d_in[14]; const float* ln2_b = (const float*)d_in[15];
  const float* W_ff1 = (const float*)d_in[16]; const float* b_ff1 = (const float*)d_in[17];
  const float* W_ff2 = (const float*)d_in[18]; const float* b_ff2 = (const float*)d_in[19];
  const float* Wg_in = (const float*)d_in[20]; const float* bg_in = (const float*)d_in[21];
  const float* Wg_mem= (const float*)d_in[22]; const float* bg_mem= (const float*)d_in[23];

  const long ND  = (long)NROWS * Dm;
  const long NF  = (long)NROWS * Fm;
  const long BSD = (long)Bm * Sm * Dm;
  const long BD  = (long)Bm * Dm;

  char* ws = (char*)d_ws;
  size_t cur = 0;
  auto alloc = [&](size_t bytes) -> char* {
    char* p = ws + cur;
    cur += (bytes + 255) & ~(size_t)255;
    return p;
  };
  float*          seq   = (float*)         alloc(ND * 4);
  unsigned short* actb  = (unsigned short*)alloc(ND * 2);       // h / ctx / h2
  float*          qkvb  = (float*)         alloc(3 * ND * 4);   // q|k|v fused, reused
  unsigned short* ffb   = (unsigned short*)alloc(NF * 2);       // relu(ff1) bf16
  float*          gmemb = (float*)         alloc((long)Bm * Sm * Gm * 4);
  float*          ginb  = (float*)         alloc((long)Bm * Gm * 4);
  unsigned short* tmemb = (unsigned short*)alloc(BSD * 2);      // tanh(memory) bf16
  unsigned short* xb    = (unsigned short*)alloc(BD * 2);
  float*          bqkv  = (float*)         alloc(QKV_N * 4);    // concat biases
  // packed weights: (K/2 x N) dwords each
  unsigned* wE    = (unsigned*)alloc((long)Dm * Dm * 2);
  unsigned* wQKV  = (unsigned*)alloc((long)Dm * QKV_N * 2);
  unsigned* wO    = (unsigned*)alloc((long)Dm * Dm * 2);
  unsigned* wF1   = (unsigned*)alloc((long)Dm * Fm * 2);
  unsigned* wF2   = (unsigned*)alloc((long)Fm * Dm * 2);
  unsigned* wGin  = (unsigned*)alloc((long)Sm * Dm * Gm * 2);
  unsigned* wGmem = (unsigned*)alloc((long)Dm * Gm * 2);
  // buffer reuse after attention:
  float* attn_out = qkvb;         // N x D
  float* encb     = qkvb + ND;    // N x D

  auto blocks = [](long n) { return (unsigned)((n + 255) / 256); };
  const long DD2 = (long)Dm * Dm / 2;

  // --- activations to bf16; weights converted + packed into K-pair dword layout ---
  cvt_bf16_kernel<<<blocks(BD), 256, 0, stream>>>(x, xb, BD, 0);
  cvt_bf16_kernel<<<blocks(BSD), 256, 0, stream>>>(memory, tmemb, BSD, 1);  // tanh
  pack_weight_kernel<<<blocks(DD2), 256, 0, stream>>>(W_embed, wE, DD2, 10, Dm, 0);
  pack_weight_kernel<<<blocks(DD2), 256, 0, stream>>>(Wq, wQKV, DD2, 10, QKV_N, 0);
  pack_weight_kernel<<<blocks(DD2), 256, 0, stream>>>(Wk, wQKV, DD2, 10, QKV_N, Dm);
  pack_weight_kernel<<<blocks(DD2), 256, 0, stream>>>(Wv, wQKV, DD2, 10, QKV_N, 2 * Dm);
  pack_weight_kernel<<<blocks(DD2), 256, 0, stream>>>(Wo, wO, DD2, 10, Dm, 0);
  pack_weight_kernel<<<blocks((long)Dm*Fm/2), 256, 0, stream>>>(W_ff1, wF1, (long)Dm*Fm/2, 12, Fm, 0);
  pack_weight_kernel<<<blocks((long)Fm*Dm/2), 256, 0, stream>>>(W_ff2, wF2, (long)Fm*Dm/2, 10, Dm, 0);
  pack_weight_kernel<<<blocks((long)Sm*Dm*Gm/2), 256, 0, stream>>>(Wg_in, wGin, (long)Sm*Dm*Gm/2, 11, Gm, 0);
  pack_weight_kernel<<<blocks((long)Dm*Gm/2), 256, 0, stream>>>(Wg_mem, wGmem, (long)Dm*Gm/2, 11, Gm, 0);
  copy_f32_kernel<<<blocks(Dm), 256, 0, stream>>>(bq, bqkv, Dm);
  copy_f32_kernel<<<blocks(Dm), 256, 0, stream>>>(bk, bqkv + Dm, Dm);
  copy_f32_kernel<<<blocks(Dm), 256, 0, stream>>>(bv, bqkv + 2 * Dm, Dm);

  // --- seq = concat(memory, x@W_embed + b_embed) ---
  mem_to_seq_kernel<<<blocks(BSD), 256, 0, stream>>>(memory, seq);
  gemm_wmma_bf16_kernel<0, false, false, Dm, Dm><<<dim3(Dm/128, Bm/128), 256, 0, stream>>>(
      xb, wE, b_embed, nullptr, seq, nullptr, /*c_base=*/8 * Dm, /*pitch=*/Tm * Dm);

  // --- LN1 -> h (bf16) ---
  layernorm_bf16_kernel<<<NROWS, 256, 0, stream>>>(seq, ln1_g, ln1_b, actb);

  // --- fused QKV GEMM (h read once) ---
  gemm_wmma_bf16_kernel<0, false, false, QKV_N, Dm><<<dim3(QKV_N/128, NROWS/128), 256, 0, stream>>>(
      actb, wQKV, bqkv, nullptr, qkvb, nullptr, 0, QKV_N);

  // --- attention (ctx written as bf16 into actb) ---
  attention_kernel<<<Bm * Hm, 64, 0, stream>>>(qkvb, actb);

  // --- attn_out = ctx@Wo + bo + seq ---
  gemm_wmma_bf16_kernel<0, true, false, Dm, Dm><<<dim3(Dm/128, NROWS/128), 256, 0, stream>>>(
      actb, wO, bo, seq, attn_out, nullptr, 0, Dm);

  // --- LN2 -> h2 (bf16) ---
  layernorm_bf16_kernel<<<NROWS, 256, 0, stream>>>(attn_out, ln2_g, ln2_b, actb);

  // --- FF1 (ReLU, bf16 out) and FF2 (+attn_out residual -> enc) ---
  gemm_wmma_bf16_kernel<1, false, true, Fm, Dm><<<dim3(Fm/128, NROWS/128), 256, 0, stream>>>(
      actb, wF1, b_ff1, nullptr, nullptr, ffb, 0, Fm);
  gemm_wmma_bf16_kernel<0, true, false, Dm, Fm><<<dim3(Dm/128, NROWS/128), 256, 0, stream>>>(
      ffb, wF2, b_ff2, attn_out, encb, nullptr, 0, Dm);

  // --- gate GEMMs: g_in = tmem.reshape(B, S*D)@Wg_in ; g_mem = tmem@Wg_mem ---
  gemm_wmma_bf16_kernel<0, false, false, Gm, Sm * Dm><<<dim3(Gm/128, Bm/128), 256, 0, stream>>>(
      tmemb, wGin, bg_in, nullptr, ginb, nullptr, 0, Gm);
  gemm_wmma_bf16_kernel<0, false, false, Gm, Dm><<<dim3(Gm/128, (Bm*Sm)/128), 256, 0, stream>>>(
      tmemb, wGmem, bg_mem, nullptr, gmemb, nullptr, 0, Gm);

  // --- gates -> next_memory, written twice (tuple (nm, nm)) ---
  gate_combine_kernel<<<blocks(BSD), 256, 0, stream>>>(
      encb, memory, gmemb, ginb, (float*)d_out);
}
